// BasicAttentionBlock_32349693673653
// MI455X (gfx1250) — matrix operand order repaired and checked by reading, not compiled, not run on hardware
//
#include <hip/hip_runtime.h>
#include <math.h>

// ---------------------------------------------------------------------------
// BasicAttentionBlock for MI455X (gfx1250):
//   - all GEMMs via v_wmma_f32_16x16x32_f16 (f32 accumulate)
//   - global->LDS staging via GLOBAL_LOAD_ASYNC_TO_LDS_B128 (ASYNCcnt) when
//     the toolchain declares the builtin, else load+ds_store fallback
// B=8, C=128, H=W=32 (HW=1024, 8192 pixels), NH=8, dkh=dvh=8
// ---------------------------------------------------------------------------

typedef __attribute__((ext_vector_type(16))) _Float16 v16h;
typedef __attribute__((ext_vector_type(8)))  _Float16 v8h;
typedef __attribute__((ext_vector_type(8)))  float    v8f;
typedef __attribute__((ext_vector_type(4)))  int      v4i;

#define BN_EPS 1e-5f
#define HWI 1024

// ---- workspace byte offsets (all 4KB aligned) ----
#define OFF_XACT  ((size_t)0)                     // f16 [8192][128]   2,097,152
#define OFF_QP    (OFF_XACT + 2097152)            // f16 [64][1024][32] (cols 8..31 zero)
#define OFF_KP    (OFF_QP   + 4194304)            // f16 [64][1024][32]
#define OFF_VT    (OFF_KP   + 4194304)            // f16 [64][16][1024] (rows 8..15 zero)
#define OFF_UNP   (OFF_VT   + 2097152)            // f16 [8192][64]
#define OFF_Y16   (OFF_UNP  + 1048576)            // f16 [8192][128]
#define OFF_WQ    (OFF_Y16  + 2097152)            // f16 [192][128]
#define OFF_WCO   (OFF_WQ   + 49152)              // f16 [9][64][128]
#define OFF_WC2   (OFF_WCO  + 147456)             // f16 [9][128][128]
#define OFF_WAT   (OFF_WC2  + 294912)             // f16 [64][64]
#define ZERO_BYTES (4194304 + 4194304 + 2097152)  // Qp+Kp+Vt contiguous

// ---------------------------------------------------------------------------
// Async global->LDS (16B granule). Guarded so compile cannot regress.
// ---------------------------------------------------------------------------
#if defined(__has_builtin)
#  if __has_builtin(__builtin_amdgcn_global_load_async_to_lds_b128)
#    define ASYNC_G2L 1
#  endif
#endif

__device__ __forceinline__ void g2l_b128(_Float16* ldst, const _Float16* gsrc) {
#ifdef ASYNC_G2L
  __builtin_amdgcn_global_load_async_to_lds_b128(
      (__attribute__((address_space(1))) v4i*)gsrc,
      (__attribute__((address_space(3))) v4i*)ldst, 0, 0);
#else
  *(uint4*)ldst = *(const uint4*)gsrc;
#endif
}
__device__ __forceinline__ void g2l_fence() {
#ifdef ASYNC_G2L
#  if __has_builtin(__builtin_amdgcn_s_wait_asynccnt)
  __builtin_amdgcn_s_wait_asynccnt(0);
#  else
  asm volatile("s_wait_asynccnt 0" ::: "memory");
#  endif
#endif
}

// ---------------------------------------------------------------------------
// WMMA fragment loaders (wave32, 16x16x32 f16).
//   A (16x32 row-major [M][K]): lane<16 -> row lane, K {0..7,16..23};
//                               lane>=16 -> row lane-16, K {8..15,24..31}
//   B (32x16) stored transposed row-major [N][K]: lane&15 = column N,
//                               K order same as A: lane<16 {0..7,16..23}, else {8..15,24..31}
//   C/D f32: VGPR r -> row (r + 8*(lane>=16)), col = lane&15
// ---------------------------------------------------------------------------
__device__ __forceinline__ v16h frag_a_row(const _Float16* rowp, int lane) {
  const _Float16* rp = rowp + ((lane >> 4) << 3);
  v8h x0 = *(const v8h*)rp;
  v8h x1 = *(const v8h*)(rp + 16);
  v16h a;
#pragma unroll
  for (int i = 0; i < 8; ++i) { a[i] = x0[i]; a[i + 8] = x1[i]; }
  return a;
}
__device__ __forceinline__ v16h frag_a(const _Float16* base, int ld, int lane) {
  return frag_a_row(base + (size_t)(lane & 15) * ld, lane);
}
__device__ __forceinline__ v16h frag_b(const _Float16* baseT, int ld, int lane) {
  return frag_a_row(baseT + (size_t)(lane & 15) * ld, lane);
}
__device__ __forceinline__ v8f wmma(v16h a, v16h b, v8f c) {
  return __builtin_amdgcn_wmma_f32_16x16x32_f16(false, a, false, b, (short)0, c,
                                                false, false);
}

// ---------------------------------------------------------------------------
// Prep kernels
// ---------------------------------------------------------------------------
__global__ __launch_bounds__(256) void k_zero(uint4* p) {
  p[blockIdx.x * 256 + threadIdx.x] = make_uint4(0u, 0u, 0u, 0u);
}

// relu(bn1(x)) -> f16 pixel-major [pix][c]
__global__ __launch_bounds__(256) void k_act(const float* __restrict__ x,
                                             const float* __restrict__ g,
                                             const float* __restrict__ bb,
                                             const float* __restrict__ m,
                                             const float* __restrict__ v,
                                             _Float16* __restrict__ act) {
  int i = blockIdx.x * 256 + threadIdx.x;        // over [8][128][1024]
  int b = i >> 17, c = (i >> 10) & 127, pin = i & 1023;
  float s = g[c] * rsqrtf(v[c] + BN_EPS);
  float val = (x[i] - m[c]) * s + bb[c];
  act[((size_t)(b * HWI + pin)) * 128 + c] = (_Float16)fmaxf(val, 0.f);
}

// pack weights to f16: qkv [192][128]; conv taps [tap][O][C]; attn [64][64]
__global__ __launch_bounds__(256) void k_packw(const float* __restrict__ wqkv,
                                               const float* __restrict__ wco,
                                               const float* __restrict__ wc2,
                                               const float* __restrict__ wat,
                                               _Float16* __restrict__ pq,
                                               _Float16* __restrict__ pco,
                                               _Float16* __restrict__ pc2,
                                               _Float16* __restrict__ pat) {
  int i = blockIdx.x * 256 + threadIdx.x;        // 0 .. 249855
  if (i < 24576) {
    pq[i] = (_Float16)wqkv[i];
  } else if (i < 24576 + 73728) {
    int j = i - 24576;
    int tap = j >> 13, o = (j >> 7) & 63, c = j & 127;
    pco[j] = (_Float16)wco[((size_t)(o * 128 + c)) * 9 + tap];
  } else if (i < 24576 + 73728 + 147456) {
    int j = i - 24576 - 73728;
    int tap = j >> 14, o = (j >> 7) & 127, c = j & 127;
    pc2[j] = (_Float16)wc2[((size_t)(o * 128 + c)) * 9 + tap];
  } else if (i < 24576 + 73728 + 147456 + 4096) {
    int j = i - 24576 - 73728 - 147456;
    pat[j] = (_Float16)wat[j];
  }
}

// ---------------------------------------------------------------------------
// qkv 1x1 conv as GEMM + scatter to padded Q (pre-scaled), padded K, V^T
// grid: 768 blocks x 8 waves = 512 mtiles x 12 ntiles
// ---------------------------------------------------------------------------
__global__ __launch_bounds__(256) void k_qkv(const _Float16* __restrict__ act,
                                             const _Float16* __restrict__ wq,
                                             const float* __restrict__ bias,
                                             _Float16* __restrict__ Qp,
                                             _Float16* __restrict__ Kp,
                                             _Float16* __restrict__ Vt) {
  int wid = blockIdx.x * 8 + (threadIdx.x >> 5);
  int lane = threadIdx.x & 31;
  int mt = wid / 12, nt = wid % 12;
  const _Float16* Ab = act + (size_t)mt * 16 * 128;
  const _Float16* Bb = wq + (size_t)nt * 16 * 128;
  v8f acc = {};
#pragma unroll
  for (int kk = 0; kk < 4; ++kk) {
    __builtin_prefetch(Ab + (kk + 1) * 32, 0, 1);
    v16h a = frag_a(Ab + kk * 32, 128, lane);
    v16h b = frag_b(Bb + kk * 32, 128, lane);
    acc = wmma(a, b, acc);
  }
  int n = nt * 16 + (lane & 15);
  float bs = bias[n];
#pragma unroll
  for (int r = 0; r < 8; ++r) {
    int m = r + ((lane >> 4) << 3);
    int pixel = mt * 16 + m;
    int b = pixel >> 10, pin = pixel & 1023;
    float val = acc[r] + bs;
    if (n < 64) {            // q (pre-scaled by dkh^-0.5)
      int hh = n >> 3, c = n & 7;
      Qp[(((size_t)(b * 8 + hh)) * HWI + pin) * 32 + c] =
          (_Float16)(val * 0.35355339059327373f);
    } else if (n < 128) {    // k
      int nn = n - 64, hh = nn >> 3, c = nn & 7;
      Kp[(((size_t)(b * 8 + hh)) * HWI + pin) * 32 + c] = (_Float16)val;
    } else {                 // v transposed [bh][c][pix]
      int nn = n - 128, hh = nn >> 3, c = nn & 7;
      Vt[((size_t)(b * 8 + hh)) * 16 * HWI + (size_t)c * HWI + pin] =
          (_Float16)val;
    }
  }
}

// ---------------------------------------------------------------------------
// Flash attention with relative position logits.
// grid: 512 blocks = (b,h) x (HW/128); 8 waves, each owns 16 q rows.
// Per 32-key step: block async-stages K(32x32)/V(16x32) tiles into LDS,
// 2 logits WMMA + rel-pos add + online softmax + 1 PV WMMA.
// ---------------------------------------------------------------------------
__global__ __launch_bounds__(256) void k_attn(const _Float16* __restrict__ Qp,
                                              const _Float16* __restrict__ Kp,
                                              const _Float16* __restrict__ Vt,
                                              const float* __restrict__ relw,
                                              const float* __restrict__ relh,
                                              _Float16* __restrict__ unproj) {
  __shared__ __align__(16) float s_ph[128 * 32];      // [lq][yk]
  __shared__ __align__(16) float s_pw[128 * 32];      // [lq][xk]
  __shared__ __align__(16) float s_rh[63 * 8];
  __shared__ __align__(16) float s_rw[63 * 8];
  __shared__ __align__(16) _Float16 s_p[8 * 16 * 32]; // per-wave P tile
  __shared__ __align__(16) _Float16 s_kt[32 * 32];    // K tile [j][c]
  __shared__ __align__(16) _Float16 s_vt[16 * 32];    // V tile [c][j]

  const int tid = threadIdx.x;
  const int lane = tid & 31;
  const int wv = tid >> 5;
  const int bh = blockIdx.x >> 3;
  const int b = bh >> 3, h = bh & 7;
  const int q0 = (blockIdx.x & 7) * 128;

  for (int i = tid; i < 63 * 8; i += 256) { s_rh[i] = relh[i]; s_rw[i] = relw[i]; }
  __syncthreads();

  const _Float16* Qb = Qp + (size_t)bh * HWI * 32;
  const _Float16* Kb = Kp + (size_t)bh * HWI * 32;
  const _Float16* Vb = Vt + (size_t)bh * 16 * HWI;

  // rel tables: ph[lq][yk] = q . rel_h[yk-yq+31];  pw[lq][xk] = q . rel_w[xk-xq+31]
  for (int it = 0; it < 32; ++it) {
    int idx = tid + it * 256;                    // 0..8191
    int sel = idx >> 12, e = idx & 4095;
    int lq = e >> 5, kk = e & 31;
    int qi = q0 + lq;
    const _Float16* qr = Qb + (size_t)qi * 32;
    const float* rr = sel ? (s_rw + (kk - (qi & 31) + 31) * 8)
                          : (s_rh + (kk - (qi >> 5) + 31) * 8);
    float acc = 0.f;
#pragma unroll
    for (int c = 0; c < 8; ++c) acc += (float)qr[c] * rr[c];
    (sel ? s_pw : s_ph)[e] = acc;
  }
  __syncthreads();

  const int qw = q0 + wv * 16;
  const int lrow = lane & 15, lhi = lane >> 4;
  v16h aQ = frag_a(Qb + (size_t)qw * 32, 32, lane);

  float m8[8], l8[8];
#pragma unroll
  for (int r = 0; r < 8; ++r) { m8[r] = -1e30f; l8[r] = 0.f; }
  v8f Oacc = {};
  _Float16* pb = s_p + wv * 512;

  for (int ks = 0; ks < 32; ++ks) {
    // cooperative async stage of K and V tiles for this 32-key step
    if (tid < 192) {
      if (tid < 128) {           // K: 32 rows x 32 halves (4 x 16B per row)
        int j = tid >> 2, c = (tid & 3) << 3;
        g2l_b128(s_kt + j * 32 + c, Kb + (size_t)(ks * 32 + j) * 32 + c);
      } else {                   // V: 16 rows x 32 halves
        int i2 = tid - 128;
        int cc = i2 >> 2, c = (i2 & 3) << 3;
        g2l_b128(s_vt + cc * 32 + c, Vb + (size_t)cc * HWI + ks * 32 + c);
      }
    }
    g2l_fence();
    __syncthreads();

    v16h b0 = frag_b(s_kt, 32, lane);
    v16h b1 = frag_b(s_kt + 16 * 32, 32, lane);
    v8f c0 = {}, c1 = {};
    c0 = wmma(aQ, b0, c0);
    c1 = wmma(aQ, b1, c1);

    float p0[8], p1[8];
#pragma unroll
    for (int r = 0; r < 8; ++r) {
      int m = r + (lhi << 3);
      int lq = wv * 16 + m;
      int j0 = ks * 32 + lrow, j1 = j0 + 16;
      p0[r] = c0[r] + s_ph[(lq << 5) + (j0 >> 5)] + s_pw[(lq << 5) + (j0 & 31)];
      p1[r] = c1[r] + s_ph[(lq << 5) + (j1 >> 5)] + s_pw[(lq << 5) + (j1 & 31)];
    }
#pragma unroll
    for (int r = 0; r < 8; ++r) {
      float t = fmaxf(p0[r], p1[r]);
#pragma unroll
      for (int off = 1; off < 16; off <<= 1) t = fmaxf(t, __shfl_xor(t, off, 16));
      float mn = fmaxf(m8[r], t);
      float sc = __expf(m8[r] - mn);
      float e0 = __expf(p0[r] - mn);
      float e1 = __expf(p1[r] - mn);
      float rs = e0 + e1;
#pragma unroll
      for (int off = 1; off < 16; off <<= 1) rs += __shfl_xor(rs, off, 16);
      l8[r] = l8[r] * sc + rs;
      m8[r] = mn;
      Oacc[r] *= sc;
      p0[r] = e0; p1[r] = e1;
    }
    // re-layout P (C fragment) -> A fragment via per-wave LDS tile
#pragma unroll
    for (int r = 0; r < 8; ++r) {
      int m = r + (lhi << 3);
      pb[m * 32 + lrow] = (_Float16)p0[r];
      pb[m * 32 + 16 + lrow] = (_Float16)p1[r];
    }
    v16h aP = frag_a(pb, 32, lane);
    v16h bV = frag_b(s_vt, 32, lane);    // rows 8..15 are zero pad
    Oacc = wmma(aP, bV, Oacc);
    __syncthreads();                     // protect s_kt/s_vt before next stage
  }

  if (lrow < 8) {
#pragma unroll
    for (int r = 0; r < 8; ++r) {
      int m = r + (lhi << 3);
      int pix = b * HWI + qw + m;
      unproj[(size_t)pix * 64 + h * 8 + lrow] = (_Float16)(Oacc[r] / l8[r]);
    }
  }
}

// ---------------------------------------------------------------------------
// 3x3 convs: one block per image row; zero-padded 3x34x128 f16 halo in LDS
// staged with async loads; waves = 2 m-halves x N/16 n-tiles.
// ---------------------------------------------------------------------------
#define HALO_PITCH (34 * 128)  // halves per halo row

__device__ __forceinline__ void stage_halo(const _Float16* __restrict__ act,
                                           _Float16* hl, int b, int y,
                                           int tid, int nthr) {
  const uint4 z = make_uint4(0u, 0u, 0u, 0u);
  for (int i = tid; i < 3 * HALO_PITCH / 8; i += nthr)   // zero 1632 x 16B
    *(uint4*)(hl + i * 8) = z;
  __syncthreads();
  for (int i = tid; i < 3 * 512; i += nthr) {            // 3 rows x 512 x 16B
    int r = i >> 9, c = i & 511;
    int sy = y - 1 + r;
    if ((unsigned)sy < 32u)
      g2l_b128(hl + r * HALO_PITCH + 128 + c * 8,
               act + ((size_t)(b * HWI + sy * 32)) * 128 + c * 8);
  }
  g2l_fence();
  __syncthreads();
}

__device__ __forceinline__ v8f conv_row_acc(const _Float16* hl,
                                            const _Float16* __restrict__ wt,
                                            int Nout, int nt, int x0, int lane) {
  v8f acc = {};
  int x = x0 + (lane & 15);
#pragma unroll
  for (int tap = 0; tap < 9; ++tap) {
    const _Float16* rowp = hl + (tap / 3) * HALO_PITCH + (x + tap % 3) * 128;
    const _Float16* wb = wt + ((size_t)tap * Nout + nt * 16) * 128;
#pragma unroll
    for (int kk = 0; kk < 4; ++kk) {
      v16h a = frag_a_row(rowp + kk * 32, lane);
      v16h bf = frag_b(wb + kk * 32, 128, lane);
      acc = wmma(a, bf, acc);
    }
  }
  return acc;
}

// convout 3x3 (N=64) + bias + bn2 + relu -> y16 channels [0,64)
// grid: 256 blocks (one per image row) x 256 threads (2 x 4 waves)
__global__ __launch_bounds__(256) void k_convout(const _Float16* __restrict__ act,
                                                 const _Float16* __restrict__ wt,
                                                 const float* __restrict__ cb,
                                                 const float* __restrict__ g2,
                                                 const float* __restrict__ b2,
                                                 const float* __restrict__ m2,
                                                 const float* __restrict__ v2,
                                                 _Float16* __restrict__ y16) {
  __shared__ __align__(16) _Float16 hl[3 * HALO_PITCH];
  int tid = threadIdx.x, lane = tid & 31, wv = tid >> 5;
  int b = blockIdx.x >> 5, y = blockIdx.x & 31;
  stage_halo(act, hl, b, y, tid, 256);
  int x0 = (wv >> 2) << 4, nt = wv & 3;
  v8f acc = conv_row_acc(hl, wt, 64, nt, x0, lane);
  int n = nt * 16 + (lane & 15);
  float s = g2[n] * rsqrtf(v2[n] + BN_EPS);
  float sh = b2[n] - m2[n] * s;
  float bs = cb[n];
#pragma unroll
  for (int r = 0; r < 8; ++r) {
    int pixel = b * HWI + y * 32 + x0 + r + ((lane >> 4) << 3);
    float val = (acc[r] + bs) * s + sh;
    y16[(size_t)pixel * 128 + n] = (_Float16)fmaxf(val, 0.f);
  }
}

// attn 1x1 projection (64->64) + bias + bn2 + relu -> y16 channels [64,128)
// grid: 256 blocks x 8 waves = 512 mtiles x 4 ntiles
__global__ __launch_bounds__(256) void k_attnproj(const _Float16* __restrict__ unp,
                                                  const _Float16* __restrict__ wt,
                                                  const float* __restrict__ ab,
                                                  const float* __restrict__ g2,
                                                  const float* __restrict__ b2,
                                                  const float* __restrict__ m2,
                                                  const float* __restrict__ v2,
                                                  _Float16* __restrict__ y16) {
  int wid = blockIdx.x * 8 + (threadIdx.x >> 5);
  int lane = threadIdx.x & 31;
  int mt = wid >> 2, nt = wid & 3;
  v8f acc = {};
#pragma unroll
  for (int kk = 0; kk < 2; ++kk) {
    v16h a = frag_a(unp + (size_t)mt * 16 * 64 + kk * 32, 64, lane);
    v16h b = frag_b(wt + (size_t)nt * 16 * 64 + kk * 32, 64, lane);
    acc = wmma(a, b, acc);
  }
  int n = nt * 16 + (lane & 15);
  int ch = 64 + n;
  float s = g2[ch] * rsqrtf(v2[ch] + BN_EPS);
  float sh = b2[ch] - m2[ch] * s;
  float bs = ab[n];
#pragma unroll
  for (int r = 0; r < 8; ++r) {
    int pixel = mt * 16 + r + ((lane >> 4) << 3);
    float val = (acc[r] + bs) * s + sh;
    y16[(size_t)pixel * 128 + ch] = (_Float16)fmaxf(val, 0.f);
  }
}

// conv2 3x3 (128->128, no bias) + residual add -> out (NCHW f32)
// grid: 256 blocks (one per image row) x 512 threads (2 x 8 waves)
__global__ __launch_bounds__(512) void k_conv2(const _Float16* __restrict__ y16,
                                               const _Float16* __restrict__ wt,
                                               const float* __restrict__ xin,
                                               float* __restrict__ out) {
  __shared__ __align__(16) _Float16 hl[3 * HALO_PITCH];
  int tid = threadIdx.x, lane = tid & 31, wv = tid >> 5;
  int b = blockIdx.x >> 5, y = blockIdx.x & 31;
  stage_halo(y16, hl, b, y, tid, 512);
  int x0 = (wv >> 3) << 4, nt = wv & 7;
  v8f acc = conv_row_acc(hl, wt, 128, nt, x0, lane);
  int n = nt * 16 + (lane & 15);
#pragma unroll
  for (int r = 0; r < 8; ++r) {
    int pin = y * 32 + x0 + r + ((lane >> 4) << 3);
    size_t oi = ((size_t)(b * 128 + n)) * HWI + pin;
    out[oi] = xin[oi] + acc[r];
  }
}

// ---------------------------------------------------------------------------
extern "C" void kernel_launch(void* const* d_in, const int* in_sizes, int n_in,
                              void* d_out, int out_size, void* d_ws, size_t ws_size,
                              hipStream_t stream) {
  const float* x     = (const float*)d_in[0];
  const float* bn1_g = (const float*)d_in[1];
  const float* bn1_b = (const float*)d_in[2];
  const float* bn1_m = (const float*)d_in[3];
  const float* bn1_v = (const float*)d_in[4];
  const float* qkv_w = (const float*)d_in[5];
  const float* qkv_b = (const float*)d_in[6];
  const float* cow_w = (const float*)d_in[7];
  const float* cow_b = (const float*)d_in[8];
  const float* att_w = (const float*)d_in[9];
  const float* att_b = (const float*)d_in[10];
  const float* rel_w = (const float*)d_in[11];
  const float* rel_h = (const float*)d_in[12];
  const float* bn2_g = (const float*)d_in[13];
  const float* bn2_b = (const float*)d_in[14];
  const float* bn2_m = (const float*)d_in[15];
  const float* bn2_v = (const float*)d_in[16];
  const float* c2_w  = (const float*)d_in[17];
  float* out = (float*)d_out;

  char* ws = (char*)d_ws;
  _Float16* xact = (_Float16*)(ws + OFF_XACT);
  _Float16* Qp   = (_Float16*)(ws + OFF_QP);
  _Float16* Kp   = (_Float16*)(ws + OFF_KP);
  _Float16* Vt   = (_Float16*)(ws + OFF_VT);
  _Float16* unp  = (_Float16*)(ws + OFF_UNP);
  _Float16* y16  = (_Float16*)(ws + OFF_Y16);
  _Float16* pwq  = (_Float16*)(ws + OFF_WQ);
  _Float16* pwco = (_Float16*)(ws + OFF_WCO);
  _Float16* pwc2 = (_Float16*)(ws + OFF_WC2);
  _Float16* pwat = (_Float16*)(ws + OFF_WAT);

  k_zero<<<ZERO_BYTES / 16 / 256, 256, 0, stream>>>((uint4*)(ws + OFF_QP));
  k_act<<<4096, 256, 0, stream>>>(x, bn1_g, bn1_b, bn1_m, bn1_v, xact);
  k_packw<<<976, 256, 0, stream>>>(qkv_w, cow_w, c2_w, att_w, pwq, pwco, pwc2, pwat);
  k_qkv<<<768, 256, 0, stream>>>(xact, pwq, qkv_b, Qp, Kp, Vt);
  k_attn<<<512, 256, 0, stream>>>(Qp, Kp, Vt, rel_w, rel_h, unp);
  k_convout<<<256, 256, 0, stream>>>(xact, pwco, cow_b, bn2_g, bn2_b, bn2_m, bn2_v, y16);
  k_attnproj<<<256, 256, 0, stream>>>(unp, pwat, att_b, bn2_g, bn2_b, bn2_m, bn2_v, y16);
  k_conv2<<<256, 512, 0, stream>>>(y16, pwc2, x, out);
}
